// MOE_73409581023887
// MI455X (gfx1250) — matrix operand order repaired and checked by reading, not compiled, not run on hardware
//
#include <hip/hip_runtime.h>
#include <hip/hip_bf16.h>

// ---------------------------------------------------------------------------
// MoE forward for gfx1250 (MI455X).
//  * Expert GEMMs + proj on WMMA matrix cores (v_wmma_f32_16x16x32_f16,
//    fp32 accumulate), 128x128x64 block tiles, 8 waves/block (wave32).
//  * Global->LDS staging uses CDNA5 async-to-LDS DMA
//    (global_load_async_to_lds_b128 + s_wait_asynccnt) with double-buffered
//    LDS, overlapping DMA of tile k+1 with WMMA on tile k.
//  * Weights are converted fp32->fp16 AND transposed to [n][k] once, so all
//    GEMM staging is contiguous b128 copies and all fragments are two
//    ds_load_b128 matching the ISA 7.12.2 lane layouts.
//  * Gating/softmax/top-2/routing are exact fp32; routing slot = stable rank
//    (identical to the reference's stable argsort + carried counts).
// ---------------------------------------------------------------------------

typedef __attribute__((ext_vector_type(16))) _Float16 v16h;
typedef __attribute__((ext_vector_type(8)))  float    v8f;

#define E_    8
#define D_    768
#define FFN_  3072
#define OUT_  768
#define NTOK  8192          // B*T*H = 4*512*4
#define CAP_  3072          // ceil(2*8192*1.5/8)
#define TH_   2048          // T*H (embedding broadcast period)

#define USE_ASYNC_STAGE 1

__device__ __forceinline__ float gelu_exact(float x) {
    return 0.5f * x * (1.0f + erff(x * 0.70710678118654752f));
}

// ---------------------------------------------------------------------------
__global__ void moe_init_kernel(int* __restrict__ slot_token, int* __restrict__ counts0) {
    int i = blockIdx.x * blockDim.x + threadIdx.x;
    if (i < E_ * CAP_) slot_token[i] = -1;
    if (i < E_)        counts0[i] = 0;
}

// fp32 -> fp16 (no transpose; proj_w is already [n][k])
__global__ void f32_to_f16_kernel(const float* __restrict__ in, _Float16* __restrict__ out, int n) {
    int i = blockIdx.x * blockDim.x + threadIdx.x;
    if (i < n) out[i] = (_Float16)in[i];
}

// fp32 [e][K][N] -> fp16 [e][N][K], LDS-tiled 64x64 transpose (coalesced both sides)
__global__ __launch_bounds__(256) void transpose_f32_to_f16_kernel(
    const float* __restrict__ in, _Float16* __restrict__ out, int K, int N)
{
    __shared__ _Float16 tile[64][66];     // stride 66 elems -> conflict-free column reads
    const float* ip = in + (size_t)blockIdx.z * K * N;
    _Float16*    op = out + (size_t)blockIdx.z * N * K;
    int n0 = blockIdx.x * 64, k0 = blockIdx.y * 64;
    int tn  = threadIdx.x & 63;
    int tk4 = threadIdx.x >> 6;           // 0..3
#pragma unroll
    for (int r = 0; r < 64; r += 4)
        tile[r + tk4][tn] = (_Float16)ip[(size_t)(k0 + r + tk4) * N + (n0 + tn)];
    __syncthreads();
#pragma unroll
    for (int r = 0; r < 64; r += 4)
        op[(size_t)(n0 + r + tk4) * K + (k0 + tn)] = tile[tn][r + tk4];
}

// ---------------------------------------------------------------------------
// xf = x + embedding; fp32 for gating + fp16 for GEMM A
// ---------------------------------------------------------------------------
__global__ void add_embed_kernel(const float* __restrict__ x, const float* __restrict__ emb,
                                 float* __restrict__ xf, _Float16* __restrict__ xh) {
    int i = blockIdx.x * blockDim.x + threadIdx.x;
    if (i >= NTOK * D_) return;
    int n = i / D_;
    int d = i - n * D_;
    int r = n % TH_;
    float v = x[i] + emb[(size_t)r * D_ + d];
    xf[i] = v;
    xh[i] = (_Float16)v;
}

// ---------------------------------------------------------------------------
// gating: one wave per token (exact fp32)
// ---------------------------------------------------------------------------
__global__ __launch_bounds__(256) void gate_kernel(
    const float* __restrict__ xf, const float* __restrict__ gw,
    int* __restrict__ top_idx, float* __restrict__ top_val,
    float* __restrict__ gates, int* __restrict__ counts0)
{
    int wid  = threadIdx.x >> 5;
    int lane = threadIdx.x & 31;
    int t = blockIdx.x * 8 + wid;
    if (t >= NTOK) return;

    float acc[E_];
#pragma unroll
    for (int e = 0; e < E_; ++e) acc[e] = 0.f;
    const float* xr = xf + (size_t)t * D_;
    for (int d = lane; d < D_; d += 32) {
        float xv = xr[d];
#pragma unroll
        for (int e = 0; e < E_; ++e) acc[e] += xv * gw[d * E_ + e];
    }
#pragma unroll
    for (int e = 0; e < E_; ++e) {
#pragma unroll
        for (int off = 16; off > 0; off >>= 1) acc[e] += __shfl_down(acc[e], off);
    }
    if (lane == 0) {
        float m = acc[0];
#pragma unroll
        for (int e = 1; e < E_; ++e) m = fmaxf(m, acc[e]);
        float p[E_]; float s = 0.f;
#pragma unroll
        for (int e = 0; e < E_; ++e) { p[e] = expf(acc[e] - m); s += p[e]; }
        float inv = 1.f / s;
#pragma unroll
        for (int e = 0; e < E_; ++e) p[e] *= inv;
        int i1 = 0; float v1 = p[0];
#pragma unroll
        for (int e = 1; e < E_; ++e) if (p[e] > v1) { v1 = p[e]; i1 = e; }  // first-max = jax tie rule
        int i2 = -1; float v2 = -1.f;
#pragma unroll
        for (int e = 0; e < E_; ++e) { if (e == i1) continue; if (p[e] > v2) { v2 = p[e]; i2 = e; } }
        float denom = fmaxf(v1 + v2, 1e-9f);
        top_idx[t * 2 + 0] = i1;  top_idx[t * 2 + 1] = i2;
        top_val[t * 2 + 0] = v1;  top_val[t * 2 + 1] = v2;
        gates[t * 2 + 0] = v1 / denom;  gates[t * 2 + 1] = v2 / denom;
        atomicAdd(&counts0[i1], 1);
    }
}

// ---------------------------------------------------------------------------
// routing rank == reference's stable argsort slot assignment
// ---------------------------------------------------------------------------
__global__ void route_rank_kernel(const int* __restrict__ top_idx, const float* __restrict__ top_val,
                                  const int* __restrict__ counts0,
                                  int* __restrict__ slot_token, int* __restrict__ token_slot)
{
    int g = blockIdx.x * blockDim.x + threadIdx.x;
    if (g >= NTOK * 2) return;
    int i = g >> 1, k = g & 1;
    int   e = top_idx[i * 2 + k];
    float v = top_val[i * 2 + k];
    int rank = 0;
    for (int j = 0; j < NTOK; ++j) {
        int   ej = top_idx[j * 2 + k];
        float vj = top_val[j * 2 + k];
        rank += (ej == e && (vj > v || (vj == v && j < i))) ? 1 : 0;
    }
    int loc = rank + (k ? counts0[e] : 0);
    if (loc < CAP_) {
        int slot = e * CAP_ + loc;
        slot_token[slot] = i;
        token_slot[g] = slot;
    } else {
        token_slot[g] = -1;
    }
}

// ---------------------------------------------------------------------------
// dispatch: buf[slot] = xh[token] (zeros for empty slots) — makes GEMM1 dense
// ---------------------------------------------------------------------------
__global__ void gather_buf_kernel(const _Float16* __restrict__ xh,
                                  const int* __restrict__ slot_token,
                                  _Float16* __restrict__ buf) {
    int i = blockIdx.x * blockDim.x + threadIdx.x;     // chunks of 8 f16
    if (i >= E_ * CAP_ * (D_ / 8)) return;
    int s  = i / (D_ / 8);
    int dc = (i - s * (D_ / 8)) * 8;
    int t = slot_token[s];
    uint4 v = make_uint4(0u, 0u, 0u, 0u);
    if (t >= 0) v = *(const uint4*)(xh + (size_t)t * D_ + dc);
    *(uint4*)(buf + (size_t)s * D_ + dc) = v;
}

// ---------------------------------------------------------------------------
// WMMA GEMM: C[M][N] = A[M][K] * B^T[N][K] (+bias, opt gelu)
// 128x128 block, BK=64 (2 wmma K-substeps), 8 waves (4x2), 32x64 per wave.
// Double-buffered LDS; async global->LDS DMA overlapped with WMMA.
// ---------------------------------------------------------------------------
constexpr int BM = 128, BN = 128, BK = 64;
constexpr int SA = BK + 8;   // 72 elems (144 B) — 16B aligned rows, padded
constexpr int SB = BK + 8;

enum { MODE_FFN1 = 0, MODE_FFN2 = 1, MODE_PROJ = 2 };

__device__ __forceinline__ v16h frag_a(const _Float16* Al, int m0, int ks, int lane) {
    // A 16x32: lane<16 -> M=lane, elems0..7=K0..7, 8..15=K16..23; lane>=16: +8/+8
    int m = lane & 15, half = lane >> 4;
    const _Float16* p = Al + (m0 + m) * SA + ks * 32;
    union { v16h v; uint4 u[2]; } r;
    r.u[0] = *(const uint4*)(p + half * 8);
    r.u[1] = *(const uint4*)(p + 16 + half * 8);
    return r.v;
}
__device__ __forceinline__ v16h frag_b(const _Float16* Bl, int n0, int ks, int lane) {
    // B 32x16: N=lane&15; lane<16 -> elems0..7=K0..7, 8..15=K16..23; lane>=16: +8/+8
    int n = lane & 15, half = lane >> 4;
    const _Float16* p = Bl + (n0 + n) * SB + ks * 32;
    union { v16h v; uint4 u[2]; } r;
    r.u[0] = *(const uint4*)(p + half * 8);
    r.u[1] = *(const uint4*)(p + 16 + half * 8);
    return r.v;
}

template <int MODE>
__global__ __launch_bounds__(256)
void wmma_gemm_kernel(const _Float16* __restrict__ A, const _Float16* __restrict__ Bt,
                      const float* __restrict__ bias_all,
                      float* __restrict__ Cf, _Float16* __restrict__ Ch,
                      int Ncols, int K)
{
    __shared__ _Float16 Al[2][BM * SA];
    __shared__ _Float16 Bl[2][BN * SB];

    const int tid   = threadIdx.x;
    const int lane  = tid & 31;
    const int wid   = tid >> 5;
    const int waveM = wid & 3;
    const int waveN = wid >> 2;
    const int blockM = blockIdx.y * BM;
    const int blockN = blockIdx.x * BN;

    const _Float16* Bg   = Bt;
    const float*    bias = bias_all;
    if (MODE != MODE_PROJ) {
        int e = blockM / CAP_;                   // CAP_ % BM == 0
        Bg   += (size_t)e * Ncols * K;           // Bt is [e][N][K]
        bias += (size_t)e * Ncols;
    }
    const _Float16* Ag  = A  + (size_t)blockM * K;
    const _Float16* Bgg = Bg + (size_t)blockN * K;

    // staging coords: 1024 chunks of 8 f16 per 128x64 tile -> 4 per thread
    int rowc[4], kcc[4];
#pragma unroll
    for (int i = 0; i < 4; ++i) { int c = tid + i * 256; rowc[i] = c >> 3; kcc[i] = (c & 7) * 8; }

    auto stage = [&](int k0, int p) {
#pragma unroll
        for (int i = 0; i < 4; ++i) {
#if USE_ASYNC_STAGE
            uint32_t la = (uint32_t)(uintptr_t)&Al[p][rowc[i] * SA + kcc[i]];
            uint64_t ga = (uint64_t)(uintptr_t)(Ag + (size_t)rowc[i] * K + k0 + kcc[i]);
            asm volatile("global_load_async_to_lds_b128 %0, %1, off" :: "v"(la), "v"(ga) : "memory");
            uint32_t lb = (uint32_t)(uintptr_t)&Bl[p][rowc[i] * SB + kcc[i]];
            uint64_t gb = (uint64_t)(uintptr_t)(Bgg + (size_t)rowc[i] * K + k0 + kcc[i]);
            asm volatile("global_load_async_to_lds_b128 %0, %1, off" :: "v"(lb), "v"(gb) : "memory");
#else
            uint4 va = *(const uint4*)(Ag + (size_t)rowc[i] * K + k0 + kcc[i]);
            *(uint4*)(&Al[p][rowc[i] * SA + kcc[i]]) = va;
            uint4 vb = *(const uint4*)(Bgg + (size_t)rowc[i] * K + k0 + kcc[i]);
            *(uint4*)(&Bl[p][rowc[i] * SB + kcc[i]]) = vb;
#endif
        }
    };

    v8f acc[2][4];
    const v8f zero8 = {0.f, 0.f, 0.f, 0.f, 0.f, 0.f, 0.f, 0.f};
#pragma unroll
    for (int mf = 0; mf < 2; ++mf)
#pragma unroll
        for (int nf = 0; nf < 4; ++nf) acc[mf][nf] = zero8;

    const int KT = K / BK;
    stage(0, 0);
    for (int kt = 0; kt < KT; ++kt) {
        const int p = kt & 1;
#if USE_ASYNC_STAGE
        asm volatile("s_wait_asynccnt 0x0" ::: "memory");
#endif
        __syncthreads();                         // all waves' DMA landed; prev reads of p^1 done
        if (kt + 1 < KT) stage((kt + 1) * BK, p ^ 1);   // DMA next tile || compute this tile
#pragma unroll
        for (int ks = 0; ks < 2; ++ks) {
            v16h af[2], bf[4];
#pragma unroll
            for (int mf = 0; mf < 2; ++mf) af[mf] = frag_a(Al[p], waveM * 32 + mf * 16, ks, lane);
#pragma unroll
            for (int nf = 0; nf < 4; ++nf) bf[nf] = frag_b(Bl[p], waveN * 64 + nf * 16, ks, lane);
#pragma unroll
            for (int mf = 0; mf < 2; ++mf)
#pragma unroll
                for (int nf = 0; nf < 4; ++nf)
                    acc[mf][nf] = __builtin_amdgcn_wmma_f32_16x16x32_f16(
                        false, af[mf], false, bf[nf], (short)0, acc[mf][nf], false, false);
        }
    }

    const int nl   = lane & 15;
    const int half = lane >> 4;
#pragma unroll
    for (int mf = 0; mf < 2; ++mf) {
#pragma unroll
        for (int nf = 0; nf < 4; ++nf) {
#pragma unroll
            for (int r = 0; r < 8; ++r) {
                int row = blockM + waveM * 32 + mf * 16 + half * 8 + r;
                int col = blockN + waveN * 64 + nf * 16 + nl;
                float vv = acc[mf][nf][r] + bias[col];
                if (MODE == MODE_FFN1) {
                    Ch[(size_t)row * Ncols + col] = (_Float16)gelu_exact(vv);
                } else {
                    Cf[(size_t)row * Ncols + col] = vv;
                }
            }
        }
    }
}

// ---------------------------------------------------------------------------
// combine: y = sum_k gate_k * eo[slot_k]; z = gelu(y) (f16 input to proj GEMM)
// ---------------------------------------------------------------------------
__global__ __launch_bounds__(256) void combine_kernel(
    const float* __restrict__ eo, const float* __restrict__ gates,
    const int* __restrict__ token_slot, _Float16* __restrict__ z)
{
    int t  = blockIdx.x;
    int s0 = token_slot[t * 2 + 0];
    int s1 = token_slot[t * 2 + 1];
    float g0 = gates[t * 2 + 0];
    float g1 = gates[t * 2 + 1];
    for (int d = threadIdx.x; d < D_; d += 256) {
        float y = 0.f;
        if (s0 >= 0) y += g0 * eo[(size_t)s0 * D_ + d];
        if (s1 >= 0) y += g1 * eo[(size_t)s1 * D_ + d];
        z[(size_t)t * D_ + d] = (_Float16)gelu_exact(y);
    }
}

// ---------------------------------------------------------------------------
extern "C" void kernel_launch(void* const* d_in, const int* in_sizes, int n_in,
                              void* d_out, int out_size, void* d_ws, size_t ws_size,
                              hipStream_t stream) {
    (void)in_sizes; (void)n_in; (void)out_size; (void)ws_size;
    const float* x      = (const float*)d_in[0];
    const float* emb    = (const float*)d_in[1];
    const float* gate_w = (const float*)d_in[2];
    const float* w1     = (const float*)d_in[3];
    const float* b1     = (const float*)d_in[4];
    const float* w2     = (const float*)d_in[5];
    const float* b2     = (const float*)d_in[6];
    const float* proj_w = (const float*)d_in[7];
    const float* proj_b = (const float*)d_in[8];
    float* out = (float*)d_out;

    size_t off = 0;
    auto alloc = [&](size_t bytes) -> void* {
        void* p = (char*)d_ws + off;
        off += (bytes + 255) & ~(size_t)255;
        return p;
    };
    float*     xf         = (float*)    alloc((size_t)NTOK * D_ * 4);
    _Float16*  xh         = (_Float16*) alloc((size_t)NTOK * D_ * 2);
    _Float16*  w1t        = (_Float16*) alloc((size_t)E_ * D_ * FFN_ * 2);   // [e][FFN][D]
    _Float16*  w2t        = (_Float16*) alloc((size_t)E_ * FFN_ * D_ * 2);   // [e][D][FFN]
    _Float16*  pwh        = (_Float16*) alloc((size_t)OUT_ * D_ * 2);        // [OUT][D] already [n][k]
    int*       top_idx    = (int*)      alloc((size_t)NTOK * 2 * 4);
    float*     top_val    = (float*)    alloc((size_t)NTOK * 2 * 4);
    float*     gatesv     = (float*)    alloc((size_t)NTOK * 2 * 4);
    int*       counts0    = (int*)      alloc((size_t)E_ * 4);
    int*       slot_token = (int*)      alloc((size_t)E_ * CAP_ * 4);
    int*       token_slot = (int*)      alloc((size_t)NTOK * 2 * 4);
    _Float16*  bufA       = (_Float16*) alloc((size_t)E_ * CAP_ * D_ * 2);
    _Float16*  h          = (_Float16*) alloc((size_t)E_ * CAP_ * FFN_ * 2);
    float*     eo         = (float*)    alloc((size_t)E_ * CAP_ * D_ * 4);
    _Float16*  z          = (_Float16*) alloc((size_t)NTOK * D_ * 2);

    const int nPW = OUT_ * D_;

    moe_init_kernel<<<(E_ * CAP_ + 255) / 256, 256, 0, stream>>>(slot_token, counts0);
    // w1 (E,D,FFN) -> w1t (E,FFN,D) ; w2 (E,FFN,D) -> w2t (E,D,FFN)
    transpose_f32_to_f16_kernel<<<dim3(FFN_ / 64, D_ / 64, E_), 256, 0, stream>>>(w1, w1t, D_, FFN_);
    transpose_f32_to_f16_kernel<<<dim3(D_ / 64, FFN_ / 64, E_), 256, 0, stream>>>(w2, w2t, FFN_, D_);
    f32_to_f16_kernel<<<(nPW + 255) / 256, 256, 0, stream>>>(proj_w, pwh, nPW);
    add_embed_kernel<<<(NTOK * D_ + 255) / 256, 256, 0, stream>>>(x, emb, xf, xh);
    gate_kernel<<<NTOK / 8, 256, 0, stream>>>(xf, gate_w, top_idx, top_val, gatesv, counts0);
    route_rank_kernel<<<(NTOK * 2 + 255) / 256, 256, 0, stream>>>(top_idx, top_val, counts0,
                                                                  slot_token, token_slot);
    gather_buf_kernel<<<(E_ * CAP_ * (D_ / 8) + 255) / 256, 256, 0, stream>>>(xh, slot_token, bufA);
    wmma_gemm_kernel<MODE_FFN1><<<dim3(FFN_ / BN, (E_ * CAP_) / BM), 256, 0, stream>>>(
        bufA, w1t, b1, nullptr, h, FFN_, D_);
    wmma_gemm_kernel<MODE_FFN2><<<dim3(D_ / BN, (E_ * CAP_) / BM), 256, 0, stream>>>(
        h, w2t, b2, eo, nullptr, D_, FFN_);
    combine_kernel<<<NTOK, 256, 0, stream>>>(eo, gatesv, token_slot, z);
    wmma_gemm_kernel<MODE_PROJ><<<dim3(OUT_ / BN, NTOK / BM), 256, 0, stream>>>(
        z, pwh, proj_b, out, nullptr, OUT_, D_);
}
